// GraphLearner_80152679678317
// MI455X (gfx1250) — compile-verified, matrix operand order checked by reading, not run on hardware
//
#include <hip/hip_runtime.h>
#include <hip/hip_bf16.h>
#include <cstdint>
#include <cstddef>

// Problem constants (match reference)
#define BATCH 16
#define NNODE 512
#define NHEAD 8
#define HDIM 64
#define FO 512
#define NNSQ (512 * 512)

typedef __attribute__((ext_vector_type(16))) __bf16 v16bf;
typedef __attribute__((ext_vector_type(16))) unsigned short v16u;
typedef __attribute__((ext_vector_type(8))) float v8f;

// float -> bf16: native convert (gfx1250 has bf16 ALU/convert support)
__device__ __forceinline__ unsigned short f2bf(float x) {
  return __builtin_bit_cast(unsigned short, (__bf16)x);
}
__device__ __forceinline__ unsigned pk2(float a, float b) {
  return (unsigned)f2bf(a) | ((unsigned)f2bf(b) << 16);
}

// ---------------------------------------------------------------------------
// Fragment-major LDS layout: a 16x32 (A) / 32x16 (B) bf16 fragment = 512 u16,
//   lane = (rc & 15) | (((kk >> 3) & 1) << 4),  idx = (kk & 7) | ((kk >> 4) << 3)
// An 8-run along kk (kk0 = 8*g) is contiguous; fragment load = 32B/lane.
// ---------------------------------------------------------------------------
__device__ __forceinline__ v16u frag_ld(const unsigned short* fragbase) {
  return *reinterpret_cast<const v16u*>(fragbase + (threadIdx.x & 31) * 16);
}
__device__ __forceinline__ int run_off(int rc, int g) {
  return ((rc & 15) | ((g & 1) << 4)) * 16 + ((g >> 1) & 1) * 8;
}

__device__ __forceinline__ v8f wmma_bf16(v16u a, v16u b, v8f c) {
  return __builtin_amdgcn_wmma_f32_16x16x32_bf16(
      false, __builtin_bit_cast(v16bf, a),
      false, __builtin_bit_cast(v16bf, b),
      (short)0, c, false, false);
}

// CDNA5 async global->LDS copy, 16B per lane (ASYNCcnt tracked)
__device__ __forceinline__ void async_b128(void* lds_dst, const void* gsrc) {
  const unsigned loff = (unsigned)(uintptr_t)lds_dst;  // low 32 bits = LDS offset
  asm volatile("global_load_async_to_lds_b128 %0, %1, off"
               :: "v"(loff), "v"(gsrc) : "memory");
}
template <int N>
__device__ __forceinline__ void async_waitN() {
  asm volatile("s_wait_asynccnt %0" :: "i"(N) : "memory");
}

// ---------------------------------------------------------------------------
// f32 -> bf16 contiguous conversion (context -> xbf)
// ---------------------------------------------------------------------------
__global__ __launch_bounds__(256) void cvt_bf16_kernel(
    const float* __restrict__ src, unsigned short* __restrict__ dst) {
  const size_t rid = (size_t)blockIdx.x * 256 + threadIdx.x;  // one 8-run each
  const float4 f0 = *reinterpret_cast<const float4*>(src + rid * 8);
  const float4 f1 = *reinterpret_cast<const float4*>(src + rid * 8 + 4);
  uint4 u;
  u.x = pk2(f0.x, f0.y); u.y = pk2(f0.z, f0.w);
  u.z = pk2(f1.x, f1.y); u.w = pk2(f1.z, f1.w);
  *reinterpret_cast<uint4*>(dst + rid * 8) = u;
}

// ---------------------------------------------------------------------------
// Weight transpose+convert: W[K][512] f32 -> WT[512][K] bf16
// ---------------------------------------------------------------------------
__global__ __launch_bounds__(256) void wconv_kernel(
    const float* __restrict__ W, unsigned short* __restrict__ WT, int K) {
  const int rid = blockIdx.x * 256 + threadIdx.x;   // runs = 512*(K/8)
  const int g8 = K >> 3;
  const int n = rid / g8, g = rid % g8;
  float e[8];
#pragma unroll
  for (int i = 0; i < 8; ++i) e[i] = W[(size_t)(g * 8 + i) * FO + n];
  uint4 u;
  u.x = pk2(e[0], e[1]); u.y = pk2(e[2], e[3]);
  u.z = pk2(e[4], e[5]); u.w = pk2(e[6], e[7]);
  *reinterpret_cast<uint4*>(WT + (size_t)n * K + g * 8) = u;
}

// ---------------------------------------------------------------------------
// GEMM: out[M,512] = A[M,K](bf16) * WT[512,K](bf16 transposed) + bias
// 256 thr = 8 waves (4x2), tile 128x128, BK=32, double-buffered async staging.
// Each wave: 2x4 fragments -> 8 WMMA per k-step.
// Optional outputs: f32 row-major, bf16 row-major, bf16 transposed [b][h][d][j].
// ---------------------------------------------------------------------------
#define GBM 128
#define GBN 128
#define GBK 32

__global__ __launch_bounds__(256) void gemm_bf16_kernel(
    const unsigned short* __restrict__ Abf, const unsigned short* __restrict__ WT,
    const float* __restrict__ bias, float* __restrict__ outF,
    unsigned short* __restrict__ outB, unsigned short* __restrict__ outBT, int K) {
  __shared__ unsigned short As[2 * 8 * 512];   // double-buffered 8 row-frags
  __shared__ unsigned short Bs[2 * 8 * 512];   // double-buffered 8 col-frags
  const int bm = blockIdx.x * GBM;
  const int bn = blockIdx.y * GBN;
  const int tid = (int)threadIdx.x;
  const int wave = tid >> 5, lane = tid & 31;
  const int wr = (wave & 3) * 32;
  const int wc = (wave >> 2) * 64;

  auto stage = [&](int k0, int buf) {
    unsigned short* Ad = As + buf * 4096;
    unsigned short* Bd = Bs + buf * 4096;
#pragma unroll
    for (int rep = 0; rep < 2; ++rep) {       // 512 runs per operand
      const int rid = tid + rep * 256;
      const int rc = rid >> 2, g = rid & 3;
      const int doff = (rc >> 4) * 512 + run_off(rc, g);
      async_b128(Ad + doff, Abf + (size_t)(bm + rc) * K + k0 + g * 8);
      async_b128(Bd + doff, WT + (size_t)(bn + rc) * K + k0 + g * 8);
    }
  };

  v8f acc[2][4] = {};
  const int nk = K / GBK;
  stage(0, 0);
  for (int i = 0; i < nk; ++i) {
    const int cur = i & 1;
    if (i + 1 < nk) { stage((i + 1) * GBK, cur ^ 1); async_waitN<4>(); }
    else            { async_waitN<0>(); }
    __syncthreads();
    const unsigned short* Ab = As + cur * 4096;
    const unsigned short* Bb = Bs + cur * 4096;
    v16u a[2], bf[4];
#pragma unroll
    for (int ri = 0; ri < 2; ++ri) a[ri] = frag_ld(Ab + ((wr >> 4) + ri) * 512);
#pragma unroll
    for (int ci = 0; ci < 4; ++ci) bf[ci] = frag_ld(Bb + ((wc >> 4) + ci) * 512);
#pragma unroll
    for (int ri = 0; ri < 2; ++ri)
#pragma unroll
      for (int ci = 0; ci < 4; ++ci)
        acc[ri][ci] = wmma_bf16(a[ri], bf[ci], acc[ri][ci]);
    __syncthreads();
  }

  const int col = lane & 15;
  const int rb = (lane >> 4) * 8;
#pragma unroll
  for (int ri = 0; ri < 2; ++ri)
#pragma unroll
    for (int ci = 0; ci < 4; ++ci) {
      const int c0 = bn + wc + ci * 16 + col;
      const int r0 = bm + wr + ri * 16 + rb;     // 8 consecutive rows
      const float bv = bias[c0];
      float v[8];
#pragma unroll
      for (int r = 0; r < 8; ++r) v[r] = acc[ri][ci][r] + bv;
      if (outF) {
#pragma unroll
        for (int r = 0; r < 8; ++r) outF[(size_t)(r0 + r) * FO + c0] = v[r];
      }
      if (outB) {
#pragma unroll
        for (int r = 0; r < 8; ++r) outB[(size_t)(r0 + r) * FO + c0] = f2bf(v[r]);
      }
      if (outBT) {  // [b][h][d][j]: ((m>>9)*512 + c0)*512 + (m&511), j-contig
        uint4 u;
        u.x = pk2(v[0], v[1]); u.y = pk2(v[2], v[3]);
        u.z = pk2(v[4], v[5]); u.w = pk2(v[6], v[7]);
        *reinterpret_cast<uint4*>(
            outBT + ((size_t)(r0 >> 9) * FO + c0) * NNODE + (r0 & 511)) = u;
      }
    }
}

// ---------------------------------------------------------------------------
// Fused attention for one (b, h, 32-row query tile). 256 thr = 8 waves.
// scores = (q.k^T + At*qe)/8 ; softmax over j ; out = alpha@v + (alpha.At)*Weh
// Double-buffered async staging of K^T and V tiles.
// ---------------------------------------------------------------------------
__global__ __launch_bounds__(256) void attn_kernel(
    const float* __restrict__ Qf, const unsigned short* __restrict__ Qb,
    const unsigned short* __restrict__ Kb, const unsigned short* __restrict__ VbT,
    const float* __restrict__ adj, const float* __restrict__ We,
    float* __restrict__ Out) {
  extern __shared__ char smem[];
  float* S = (float*)smem;                          // [32][512]
  float* red = S + 32 * 512;                        // [32][8]
  float* red2 = red + 256;                          // [32][8]
  float* qe = red2 + 256;                           // [32]
  float* sumv = qe + 32;                            // [32]
  float* aev = sumv + 32;                           // [32]
  unsigned short* P = (unsigned short*)(aev + 32);  // 32 frags (2 x 16) x 512
  unsigned short* qs = P + 32 * 512;                // 4 frags (2 x 2) x 512
  unsigned short* kv = qs + 4 * 512;                // 2 x 16 frags (Kt / V)

  const int b = blockIdx.x >> 3;
  const int h = blockIdx.x & 7;
  const int i0 = blockIdx.y * 32;
  const int tid = (int)threadIdx.x;
  const int wave = tid >> 5;
  const int lane = tid & 31;

  // --- stage q tile: 256 runs, one per thread (A layout: frag = fr*2+fk) ---
  {
    const int row = tid >> 3, g = tid & 7;
    async_b128(qs + ((row >> 4) * 2 + (g >> 2)) * 512 + run_off(row, g),
               Qb + ((size_t)(b * NNODE + i0 + row)) * FO + h * HDIM + g * 8);
  }
  // qe[i] = q_i . Weh  (fp32)
  if (tid < 32) {
    float s = 0.f;
    const float* qrow = Qf + ((size_t)(b * NNODE + i0 + tid)) * FO + h * HDIM;
#pragma unroll 8
    for (int d = 0; d < 64; ++d) s += qrow[d] * We[h * HDIM + d];
    qe[tid] = s;
  }

  auto stage_kt = [&](int j0, int buf) {            // B layout: frag = fk*8+fj
    unsigned short* dst = kv + buf * 8192;
#pragma unroll
    for (int t = 0; t < 4; ++t) {
      const int rid = tid + t * 256;
      const int j = rid >> 3, g = rid & 7;
      async_b128(dst + ((g >> 2) * 8 + (j >> 4)) * 512 + run_off(j, g),
                 Kb + ((size_t)(b * NNODE + j0 + j)) * FO + h * HDIM + g * 8);
    }
  };
  auto stage_v = [&](int j0, int buf) {             // B layout: kk=j, c=d
    unsigned short* dst = kv + buf * 8192;
#pragma unroll
    for (int t = 0; t < 4; ++t) {
      const int rid = tid + t * 256;
      const int dd = rid & 63, m = rid >> 6;        // m: 8-run of j
      async_b128(dst + ((m >> 2) * 4 + (dd >> 4)) * 512 + run_off(dd, m),
                 VbT + ((size_t)(b * NHEAD + h) * HDIM + dd) * NNODE + j0 + m * 8);
    }
  };

  // ---- score phase: S[32][512] via WMMA q @ k^T, pipelined tiles ----
  const int jsub = wave * 16;
  stage_kt(0, 0);
  for (int jt = 0; jt < 4; ++jt) {
    const int j0 = jt * 128;
    if (jt < 3) { stage_kt(j0 + 128, (jt + 1) & 1); async_waitN<4>(); }
    else        { async_waitN<0>(); }
    __syncthreads();
    const unsigned short* kt = kv + (jt & 1) * 8192;
    v8f acc0 = {}, acc1 = {};
#pragma unroll
    for (int ks = 0; ks < 64; ks += 32) {
      const v16u bfrag = frag_ld(kt + ((ks >> 5) * 8 + wave) * 512);
      acc0 = wmma_bf16(frag_ld(qs + (0 + (ks >> 5)) * 512), bfrag, acc0);
      acc1 = wmma_bf16(frag_ld(qs + (2 + (ks >> 5)) * 512), bfrag, acc1);
    }
    const int jc = j0 + jsub + (lane & 15);
    const int rb = (lane >> 4) * 8;
    const float* atc = adj + (size_t)b * NNSQ + (size_t)jc * NNODE + i0;  // At[b,i,j]=adj[b,j,i]
#pragma unroll
    for (int r = 0; r < 8; ++r) {
      const int row = rb + r;
      S[row * 512 + jc] = (acc0[r] + atc[row] * qe[row]) * 0.125f;
      const int row2 = 16 + rb + r;
      S[row2 * 512 + jc] = (acc1[r] + atc[row2] * qe[row2]) * 0.125f;
    }
    __syncthreads();
  }

  // ---- softmax (unnormalized; sums divided out in epilogue) ----
  {
    const int row = tid >> 3, part = tid & 7;
    float* srow = S + row * 512;
    float mx = -3.0e38f;
    for (int c = part * 64; c < part * 64 + 64; ++c) mx = fmaxf(mx, srow[c]);
    red[row * 8 + part] = mx;
    __syncthreads();
    mx = red[row * 8];
#pragma unroll
    for (int t = 1; t < 8; ++t) mx = fmaxf(mx, red[row * 8 + t]);
    __syncthreads();
    float ps = 0.f, pa = 0.f;
    const float* atp = adj + (size_t)b * NNSQ + (i0 + row);
#pragma unroll
    for (int t = 0; t < 8; ++t) {            // 8 packed runs of 8
      const int c0 = part * 64 + t * 8;
      float e[8];
#pragma unroll
      for (int u = 0; u < 8; ++u) {
        const float ee = __expf(srow[c0 + u] - mx);
        e[u] = ee; ps += ee;
        pa += ee * atp[(size_t)(c0 + u) * NNODE];
      }
      uint4 pk;
      pk.x = pk2(e[0], e[1]); pk.y = pk2(e[2], e[3]);
      pk.z = pk2(e[4], e[5]); pk.w = pk2(e[6], e[7]);
      // P as A operand: frag = fr*16 + fj, kk = c&31
      const int dst = ((row >> 4) * 16 + (c0 >> 5)) * 512 + run_off(row, c0 >> 3);
      *reinterpret_cast<uint4*>(P + dst) = pk;
    }
    red[row * 8 + part] = ps;
    red2[row * 8 + part] = pa;
    __syncthreads();
    if (part == 0) {
      float s = 0.f, a = 0.f;
#pragma unroll
      for (int t = 0; t < 8; ++t) { s += red[row * 8 + t]; a += red2[row * 8 + t]; }
      sumv[row] = s; aev[row] = a;
    }
  }

  // ---- P @ V : each wave owns one 16x16 output fragment; pipelined tiles ----
  const int rw = wave & 1, cw = wave >> 1;
  v8f acc = {};
  stage_v(0, 0);
  for (int jt = 0; jt < 4; ++jt) {
    const int j0 = jt * 128;
    if (jt < 3) { stage_v(j0 + 128, (jt + 1) & 1); async_waitN<4>(); }
    else        { async_waitN<0>(); }
    __syncthreads();
    const unsigned short* vt = kv + (jt & 1) * 8192;
#pragma unroll
    for (int kk = 0; kk < 128; kk += 32) {
      acc = wmma_bf16(frag_ld(P + (rw * 16 + ((j0 + kk) >> 5)) * 512),
                      frag_ld(vt + ((kk >> 5) * 4 + cw) * 512), acc);
    }
    __syncthreads();
  }
  {
    const int col = cw * 16 + (lane & 15);
    const int rb = (lane >> 4) * 8;
    const float we = We[h * HDIM + col];
#pragma unroll
    for (int r = 0; r < 8; ++r) {
      const int row = rw * 16 + rb + r;
      const float inv = 1.f / sumv[row];
      Out[((size_t)(b * NNODE + i0 + row)) * FO + h * HDIM + col] =
          acc[r] * inv + aev[row] * inv * we;
    }
  }
}

// ---------------------------------------------------------------------------
// Gated beta-skip: g = sigmoid([o, r, o-r].Wb); y = act(g*r + (1-g)*o)
// One wave32 per row; each lane owns 16 contiguous columns.
// ---------------------------------------------------------------------------
__global__ __launch_bounds__(256) void beta_skip_kernel(
    const float* __restrict__ O, const float* __restrict__ R,
    const float* __restrict__ Wb, unsigned short* __restrict__ Ybf,
    float* __restrict__ Yf, int final_sigmoid) {
  const int row = blockIdx.x * 8 + ((int)threadIdx.x >> 5);
  const int lane = (int)threadIdx.x & 31;
  const int c0 = lane * 16;
  const float* o = O + (size_t)row * FO + c0;
  const float* rr = R + (size_t)row * FO + c0;
  float ov[16], rv[16];
  float z = 0.f;
#pragma unroll
  for (int q = 0; q < 4; ++q) {
    const float4 fo4 = *reinterpret_cast<const float4*>(o + q * 4);
    const float4 fr4 = *reinterpret_cast<const float4*>(rr + q * 4);
    const float4 w0 = *reinterpret_cast<const float4*>(Wb + c0 + q * 4);
    const float4 w1 = *reinterpret_cast<const float4*>(Wb + FO + c0 + q * 4);
    const float4 w2 = *reinterpret_cast<const float4*>(Wb + 2 * FO + c0 + q * 4);
    ov[q*4+0] = fo4.x; ov[q*4+1] = fo4.y; ov[q*4+2] = fo4.z; ov[q*4+3] = fo4.w;
    rv[q*4+0] = fr4.x; rv[q*4+1] = fr4.y; rv[q*4+2] = fr4.z; rv[q*4+3] = fr4.w;
    z += fo4.x*w0.x + fo4.y*w0.y + fo4.z*w0.z + fo4.w*w0.w;
    z += fr4.x*w1.x + fr4.y*w1.y + fr4.z*w1.z + fr4.w*w1.w;
    z += (fo4.x-fr4.x)*w2.x + (fo4.y-fr4.y)*w2.y + (fo4.z-fr4.z)*w2.z + (fo4.w-fr4.w)*w2.w;
  }
#pragma unroll
  for (int off = 16; off > 0; off >>= 1) z += __shfl_xor(z, off, 32);
  const float g = 1.f / (1.f + __expf(-z));
  float y[16];
#pragma unroll
  for (int i = 0; i < 16; ++i) {
    float t = g * rv[i] + (1.f - g) * ov[i];
    y[i] = final_sigmoid ? (1.f / (1.f + __expf(-t))) : fmaxf(t, 0.f);
  }
  if (final_sigmoid) {
#pragma unroll
    for (int q = 0; q < 4; ++q) {
      float4 f; f.x = y[q*4]; f.y = y[q*4+1]; f.z = y[q*4+2]; f.w = y[q*4+3];
      *reinterpret_cast<float4*>(Yf + (size_t)row * FO + c0 + q * 4) = f;
    }
  } else {
#pragma unroll
    for (int q = 0; q < 2; ++q) {
      uint4 u;
      u.x = pk2(y[q*8+0], y[q*8+1]); u.y = pk2(y[q*8+2], y[q*8+3]);
      u.z = pk2(y[q*8+4], y[q*8+5]); u.w = pk2(y[q*8+6], y[q*8+7]);
      *reinterpret_cast<uint4*>(Ybf + (size_t)row * FO + c0 + q * 8) = u;
    }
  }
}

// ---------------------------------------------------------------------------
extern "C" void kernel_launch(void* const* d_in, const int* in_sizes, int n_in,
                              void* d_out, int out_size, void* d_ws, size_t ws_size,
                              hipStream_t stream) {
  (void)in_sizes; (void)n_in; (void)out_size; (void)ws_size;
  const float* context = (const float*)d_in[0];
  const float* adj = (const float*)d_in[1];

  const size_t MAT = (size_t)BATCH * NNODE * FO;  // 8192*512
  float* qf = (float*)d_ws;
  float* rf = qf + MAT;
  float* of = rf + MAT;
  unsigned short* xbf = (unsigned short*)(of + MAT);
  unsigned short* qbf = xbf + MAT;
  unsigned short* kbf = qbf + MAT;
  unsigned short* vbT = kbf + MAT;
  unsigned short* wt[4];
  wt[0] = vbT + MAT;
  wt[1] = wt[0] + (size_t)FO * FO;
  wt[2] = wt[1] + (size_t)FO * FO;
  wt[3] = wt[2] + (size_t)FO * FO;

  // attention dynamic LDS: S + reductions (f32) + P + qs + double-buffered kv
  const size_t attn_smem =
      (32 * 512 + 256 + 256 + 32 + 32 + 32) * sizeof(float) +
      (32 * 512 + 4 * 512 + 2 * 16 * 512) * sizeof(unsigned short);  // 137600 B

  const dim3 ggrid(BATCH * NNODE / GBM, FO / GBN);  // (64, 4)
  const dim3 agrid(BATCH * NHEAD, NNODE / 32);      // (128, 16)

  // context (f32) -> xbf (bf16, stride 256)
  cvt_bf16_kernel<<<(BATCH * NNODE * 256 / 8) / 256, 256, 0, stream>>>(context, xbf);

  for (int l = 0; l < 4; ++l) {
    const int base = 2 + 10 * l;
    const float* Wq = (const float*)d_in[base + 0];
    const float* bq = (const float*)d_in[base + 1];
    const float* Wk = (const float*)d_in[base + 2];
    const float* bk = (const float*)d_in[base + 3];
    const float* Wv = (const float*)d_in[base + 4];
    const float* bv = (const float*)d_in[base + 5];
    const float* We = (const float*)d_in[base + 6];
    const float* Ws = (const float*)d_in[base + 7];
    const float* bs = (const float*)d_in[base + 8];
    const float* Wb = (const float*)d_in[base + 9];
    const int K = (l == 0) ? 256 : 512;
    const int wblocks = (FO * (K / 8)) / 256;

    wconv_kernel<<<wblocks, 256, 0, stream>>>(Wq, wt[0], K);
    wconv_kernel<<<wblocks, 256, 0, stream>>>(Wk, wt[1], K);
    wconv_kernel<<<wblocks, 256, 0, stream>>>(Wv, wt[2], K);
    wconv_kernel<<<wblocks, 256, 0, stream>>>(Ws, wt[3], K);

    gemm_bf16_kernel<<<ggrid, 256, 0, stream>>>(xbf, wt[0], bq, qf, qbf, nullptr, K);
    gemm_bf16_kernel<<<ggrid, 256, 0, stream>>>(xbf, wt[1], bk, nullptr, kbf, nullptr, K);
    gemm_bf16_kernel<<<ggrid, 256, 0, stream>>>(xbf, wt[2], bv, nullptr, nullptr, vbT, K);
    gemm_bf16_kernel<<<ggrid, 256, 0, stream>>>(xbf, wt[3], bs, rf, nullptr, nullptr, K);

    attn_kernel<<<agrid, 256, attn_smem, stream>>>(qf, qbf, kbf, vbT, adj, We, of);

    beta_skip_kernel<<<BATCH * NNODE / 8, 256, 0, stream>>>(
        of, rf, Wb, xbf, (float*)d_out, (l == 3) ? 1 : 0);
  }
}
